// Qwen2GQA_128849019506
// MI455X (gfx1250) — matrix operand order repaired and checked by reading, not compiled, not run on hardware
//
#include <hip/hip_runtime.h>
#include <cstdint>

// ---------------- problem constants ----------------
#define B_    2
#define S_    2048
#define D_    3584
#define H_    28
#define KV_   4
#define HD_   128
#define NREP_ (H_ / KV_)
#define MROWS (B_ * S_)          // 4096
#define SCALE_F 0.08838834764831845f   // 128^-0.5

typedef __bf16 bf16;
typedef bf16  v16bf __attribute__((ext_vector_type(16)));
typedef float v8f   __attribute__((ext_vector_type(8)));
typedef uint32_t u32;

union Frag16 { uint4 q[2]; v16bf v; };

__device__ inline v8f wmma_bf16f32(v16bf a, v16bf b, v8f c) {
  // D = A(16x32 bf16) * B(32x16 bf16) + C(16x16 f32)
  return __builtin_amdgcn_wmma_f32_16x16x32_bf16(
      /*neg_a=*/false, a, /*neg_b=*/false, b,
      /*c_mod=*/(short)0, c, /*reuse_a=*/false, /*reuse_b=*/false);
}

// CDNA5 async global->LDS copy (ASYNCcnt-tracked), 16B per lane.
__device__ inline void async_b128(u32 lds_off, const void* base, u32 goff) {
  asm volatile("global_load_async_to_lds_b128 %0, %1, %2"
               :: "v"(lds_off), "v"(goff), "s"(base) : "memory");
}
template<int N>
__device__ inline void wait_async_le() {
  asm volatile("s_wait_asynccnt %0" :: "i"(N) : "memory");
}

// ---------------- fp32 -> bf16 conversion ----------------
__global__ __launch_bounds__(256) void cvt_f32_bf16(
    const float* __restrict__ in, bf16* __restrict__ out, int n4) {
  int i = blockIdx.x * 256 + threadIdx.x;
  if (i < n4) {
    float4 f = ((const float4*)in)[i];
    union { bf16 e[4]; uint2 u; } p;
    p.e[0] = (bf16)f.x; p.e[1] = (bf16)f.y;
    p.e[2] = (bf16)f.z; p.e[3] = (bf16)f.w;
    ((uint2*)out)[i] = p.u;
  }
}

// ---------------- bf16 WMMA GEMM:  C[M,N] = A[M,K] @ W[N,K]^T (+bias) ------
// Block 128x64, 8 waves in 4x2 grid, each wave 32x32 (2x2 WMMA C tiles).
// BK=64, double-buffered LDS, async global->LDS pipeline (ASYNCcnt).
template<bool BF16_OUT, bool HAS_BIAS>
__global__ __launch_bounds__(256) void gemm_bf16_wmma(
    const bf16* __restrict__ A, const bf16* __restrict__ W,
    const float* __restrict__ bias, void* __restrict__ Out,
    int M, int N, int K)
{
  constexpr int BM = 128, BN = 64, BK = 64, SA = 72, SB = 72; // 72-elem stride keeps 16B align
  __shared__ __align__(16) bf16 sA[2][BM * SA];
  __shared__ __align__(16) bf16 sB[2][BN * SB];
  const int tid = threadIdx.x, lane = tid & 31, wvid = tid >> 5;
  const int wm = wvid >> 1, wn = wvid & 1;
  const int half = lane >> 4, lm = lane & 15;
  const int bm0 = blockIdx.y * BM, bn0 = blockIdx.x * BN;
  const u32 sAb[2] = { (u32)(uintptr_t)sA[0], (u32)(uintptr_t)sA[1] };
  const u32 sBb[2] = { (u32)(uintptr_t)sB[0], (u32)(uintptr_t)sB[1] };
  v8f acc[2][2] = {};

  auto issue_tile = [&](int k0, int buf) {
    // A tile: 128 rows x 64 bf16 = 1024 x 16B chunks; 4/thread
#pragma unroll
    for (int i = 0; i < 4; ++i) {
      int id = tid + i * 256;
      int r = id >> 3, c = id & 7;
      async_b128(sAb[buf] + (u32)(r * SA + c * 8) * 2, A,
                 (u32)((bm0 + r) * K + k0 + c * 8) * 2);
    }
    // W tile: 64 rows x 64 bf16 = 512 chunks; 2/thread
#pragma unroll
    for (int i = 0; i < 2; ++i) {
      int id = tid + i * 256;
      int r = id >> 3, c = id & 7;
      async_b128(sBb[buf] + (u32)(r * SB + c * 8) * 2, W,
                 (u32)((bn0 + r) * K + k0 + c * 8) * 2);
    }
  };

  const int nT = K / BK;
  issue_tile(0, 0);
  for (int t = 0; t < nT; ++t) {
    const int cur = t & 1;
    if (t + 1 < nT) {
      issue_tile((t + 1) * BK, cur ^ 1);
      wait_async_le<6>();           // this wave's 6 ops for tile t are done (in-order)
    } else {
      wait_async_le<0>();
    }
    __syncthreads();                // all waves' tile-t loads complete
    const bf16* cA = sA[cur];
    const bf16* cB = sB[cur];
    Frag16 af[2][2], wf[2][2];
#pragma unroll
    for (int i = 0; i < 2; ++i) {   // A 16x32 frags, two K-chunks
      int row = wm * 32 + i * 16 + lm;
#pragma unroll
      for (int kc = 0; kc < 2; ++kc) {
        af[i][kc].q[0] = *(const uint4*)&cA[row * SA + kc * 32 + half * 8];
        af[i][kc].q[1] = *(const uint4*)&cA[row * SA + kc * 32 + 16 + half * 8];
      }
    }
#pragma unroll
    for (int j = 0; j < 2; ++j) {   // B 32x16 frags
      int col = wn * 32 + j * 16 + lm;
#pragma unroll
      for (int kc = 0; kc < 2; ++kc) {
        wf[j][kc].q[0] = *(const uint4*)&cB[col * SB + kc * 32 + half * 8];
        wf[j][kc].q[1] = *(const uint4*)&cB[col * SB + kc * 32 + 16 + half * 8];
      }
    }
#pragma unroll
    for (int kc = 0; kc < 2; ++kc)
#pragma unroll
      for (int i = 0; i < 2; ++i)
#pragma unroll
        for (int j = 0; j < 2; ++j)
          acc[i][j] = wmma_bf16f32(af[i][kc].v, wf[j][kc].v, acc[i][j]);
    __syncthreads();                // tile-t buffer free for reuse
  }

  // branch-free epilogue: C layout N across lanes, M = half*8 + vgpr
#pragma unroll
  for (int i = 0; i < 2; ++i)
#pragma unroll
    for (int j = 0; j < 2; ++j) {
      const int c = bn0 + wn * 32 + j * 16 + lm;
      const int r0 = bm0 + wm * 32 + i * 16 + half * 8;
      float bv = 0.0f;
      if constexpr (HAS_BIAS) bv = bias[c];
      if constexpr (BF16_OUT) {
        bf16* p = (bf16*)Out + (size_t)r0 * N + c;
#pragma unroll
        for (int v = 0; v < 8; ++v) { *p = (bf16)(acc[i][j][v] + bv); p += N; }
      } else {
        float* p = (float*)Out + (size_t)r0 * N + c;
#pragma unroll
        for (int v = 0; v < 8; ++v) { *p = acc[i][j][v] + bv; p += N; }
      }
    }
}

// ---------------- RoPE: [B,S,heads,HD] -> [B,heads,S,HD] ----------------
__global__ __launch_bounds__(256) void rope_kernel(
    const bf16* __restrict__ in, bf16* __restrict__ out,
    const float* __restrict__ fc, const float* __restrict__ fs,
    const int* __restrict__ start_pos, int heads)
{
  int idx = blockIdx.x * 256 + threadIdx.x;   // over B*S*heads*(HD/2)
  int hd2 = idx & 63;
  int t = idx >> 6;
  int hh = t % heads; t /= heads;
  int s = t % S_; int bb = t / S_;
  int sp = *start_pos + s;
  float c = fc[sp * 64 + hd2], sn = fs[sp * 64 + hd2];
  size_t iof = ((size_t)(bb * S_ + s) * heads + hh) * HD_ + 2 * hd2;
  float xr = (float)in[iof], xi = (float)in[iof + 1];
  size_t oof = ((size_t)(bb * heads + hh) * S_ + s) * HD_ + 2 * hd2;
  out[oof]     = (bf16)(xr * c - xi * sn);
  out[oof + 1] = (bf16)(xr * sn + xi * c);
}

// ---------------- V layout change: [B,S,KV,HD] -> [B,KV,S,HD] ------------
__global__ __launch_bounds__(256) void vtrans_kernel(
    const bf16* __restrict__ in, bf16* __restrict__ out)
{
  int idx = blockIdx.x * 256 + threadIdx.x;  // over B*S*KV*HD
  int hd = idx % HD_; int t = idx / HD_;
  int kv = t % KV_; t /= KV_;
  int s = t % S_; int b = t / S_;
  out[((size_t)(b * KV_ + kv) * S_ + s) * HD_ + hd] = in[idx];
}

// ---------------- Flash attention (causal, GQA) ----------------
// One block = one (b, h, 128 q-rows). 8 waves, each owns 16 q rows.
// K staged via async-to-LDS; V staged transposed; online softmax; WMMA QK^T & PV.
__global__ __launch_bounds__(256) void flash_attn(
    const bf16* __restrict__ Qt,   // [B,H,S,HD]
    const bf16* __restrict__ Kt,   // [B,KV,S,HD]
    const bf16* __restrict__ Vt,   // [B,KV,S,HD]
    bf16* __restrict__ O)          // [B,S,H*HD]
{
  __shared__ __align__(16) bf16 sK[32 * 144];      // [key][hd], stride 144
  __shared__ __align__(16) bf16 sV[128 * 48];      // transposed: [hd][key], stride 48
  __shared__ __align__(16) bf16 sP[8][16 * 48];    // per-wave P scratch
  const int qblk = blockIdx.x, b = blockIdx.y, h = blockIdx.z;
  const int kvh = h / NREP_;
  const int tid = threadIdx.x, lane = tid & 31, wave = tid >> 5;
  const int half = lane >> 4, lm = lane & 15;
  const int q0 = qblk * 128 + wave * 16;
  const bf16* qbase = Qt + (((size_t)b * H_ + h) * S_ + q0) * HD_;
  const bf16* kbase = Kt + ((size_t)b * KV_ + kvh) * S_ * HD_;
  const bf16* vbase = Vt + ((size_t)b * KV_ + kvh) * S_ * HD_;
  const u32 sKb = (u32)(uintptr_t)sK;

  // Q fragments: 4 HD-chunks of 32, A-layout, straight from global (16B loads)
  Frag16 qf[4];
#pragma unroll
  for (int c = 0; c < 4; ++c) {
    qf[c].q[0] = *(const uint4*)(qbase + lm * HD_ + c * 32 + half * 8);
    qf[c].q[1] = *(const uint4*)(qbase + lm * HD_ + c * 32 + 16 + half * 8);
  }
  v8f o[8] = {};
  float mrow[8], lrow[8];
#pragma unroll
  for (int v = 0; v < 8; ++v) { mrow[v] = -3.0e38f; lrow[v] = 0.0f; }

  const int nkb = (qblk + 1) * 4;   // key blocks of 32 up to causal limit of block
  for (int kb = 0; kb < nkb; ++kb) {
    const int k0 = kb * 32;
    __syncthreads();
    // stage K (32x128 bf16) via async copy: 512 chunks of 16B
#pragma unroll
    for (int i = 0; i < 2; ++i) {
      int id = tid + i * 256;
      int r = id >> 4, c = id & 15;
      async_b128(sKb + (u32)(r * 144 + c * 8) * 2, kbase,
                 (u32)((k0 + r) * HD_ + c * 8) * 2);
    }
    // stage V transposed: read 16B along hd, scatter into [hd][key]
#pragma unroll
    for (int i = 0; i < 2; ++i) {
      int id = tid + i * 256;
      int r = id >> 4, c = id & 15;
      union { uint4 u; bf16 e[8]; } d;
      d.u = *(const uint4*)(vbase + (size_t)(k0 + r) * HD_ + c * 8);
#pragma unroll
      for (int e = 0; e < 8; ++e) sV[(c * 8 + e) * 48 + r] = d.e[e];
    }
    wait_async_le<0>();
    __syncthreads();

    if (k0 <= q0 + 15) {           // this wave has unmasked keys in the block
      // ---- QK^T: scores 16 x 32 (two 16x16 C tiles) ----
      v8f sc[2] = {};
#pragma unroll
      for (int t = 0; t < 2; ++t) {
        int krow = t * 16 + lm;
#pragma unroll
        for (int c = 0; c < 4; ++c) {
          Frag16 kf;
          kf.q[0] = *(const uint4*)&sK[krow * 144 + c * 32 + half * 8];
          kf.q[1] = *(const uint4*)&sK[krow * 144 + c * 32 + 16 + half * 8];
          sc[t] = wmma_bf16f32(qf[c].v, kf.v, sc[t]);
        }
      }
      // ---- scale + causal mask (in-register) ----
#pragma unroll
      for (int t = 0; t < 2; ++t)
#pragma unroll
        for (int v = 0; v < 8; ++v) {
          int qr = q0 + half * 8 + v;
          int kc = k0 + t * 16 + lm;
          float x = sc[t][v] * SCALE_F;
          sc[t][v] = (kc > qr) ? (x - 1.0e9f) : x;
        }
      // ---- online softmax: row stats across the 16-lane half-groups ----
      float sc_old[8];
#pragma unroll
      for (int v = 0; v < 8; ++v) {
        float rm = fmaxf(sc[0][v], sc[1][v]);
#pragma unroll
        for (int off = 1; off < 16; off <<= 1)
          rm = fmaxf(rm, __shfl_xor(rm, off, 32));
        float mn = fmaxf(mrow[v], rm);
        sc_old[v] = __expf(mrow[v] - mn);
        mrow[v] = mn;
      }
#pragma unroll
      for (int t = 0; t < 2; ++t)
#pragma unroll
        for (int v = 0; v < 8; ++v)
          sc[t][v] = __expf(sc[t][v] - mrow[v]);
#pragma unroll
      for (int v = 0; v < 8; ++v) {
        float rs = sc[0][v] + sc[1][v];
#pragma unroll
        for (int off = 1; off < 16; off <<= 1)
          rs += __shfl_xor(rs, off, 32);
        lrow[v] = lrow[v] * sc_old[v] + rs;
      }
#pragma unroll
      for (int n = 0; n < 8; ++n)
#pragma unroll
        for (int v = 0; v < 8; ++v)
          o[n][v] *= sc_old[v];
      // ---- P: C-layout -> A-layout via wave-private LDS ----
      bf16* myP = sP[wave];
#pragma unroll
      for (int t = 0; t < 2; ++t)
#pragma unroll
        for (int v = 0; v < 8; ++v)
          myP[(half * 8 + v) * 48 + t * 16 + lm] = (bf16)sc[t][v];
      Frag16 pf;
      pf.q[0] = *(const uint4*)&myP[lm * 48 + half * 8];
      pf.q[1] = *(const uint4*)&myP[lm * 48 + 16 + half * 8];
      // ---- PV: O(16x128) += P(16x32) @ V(32x128) ----
#pragma unroll
      for (int n = 0; n < 8; ++n) {
        Frag16 vf;
        int hdc = n * 16 + lm;
        vf.q[0] = *(const uint4*)&sV[hdc * 48 + half * 8];
        vf.q[1] = *(const uint4*)&sV[hdc * 48 + 16 + half * 8];
        o[n] = wmma_bf16f32(pf.v, vf.v, o[n]);
      }
    }
  }
  // ---- normalize + store to [B,S,H*HD] bf16 (pointer-incremented) ----
#pragma unroll
  for (int n = 0; n < 8; ++n) {
    bf16* p = O + (size_t)(b * S_ + q0 + half * 8) * (H_ * HD_) + h * HD_ + n * 16 + lm;
#pragma unroll
    for (int v = 0; v < 8; ++v) {
      *p = (bf16)(o[n][v] / lrow[v]);
      p += H_ * HD_;
    }
  }
}

// ---------------- host-side orchestration ----------------
extern "C" void kernel_launch(void* const* d_in, const int* in_sizes, int n_in,
                              void* d_out, int out_size, void* d_ws, size_t ws_size,
                              hipStream_t stream)
{
  (void)in_sizes; (void)n_in; (void)out_size; (void)ws_size;
  const float* x     = (const float*)d_in[0];
  const float* wq_w  = (const float*)d_in[1];
  const float* wq_b  = (const float*)d_in[2];
  const float* wk_w  = (const float*)d_in[3];
  const float* wk_b  = (const float*)d_in[4];
  const float* wv_w  = (const float*)d_in[5];
  const float* wv_b  = (const float*)d_in[6];
  const float* wo_w  = (const float*)d_in[7];
  const float* fcos  = (const float*)d_in[10];
  const float* fsin  = (const float*)d_in[11];
  const int*   spos  = (const int*)d_in[13];

  char* ws = (char*)d_ws;
  size_t off = 0;
  auto alloc = [&](size_t bytes) -> char* {
    char* p = ws + off; off += (bytes + 255) & ~(size_t)255; return p;
  };
  bf16* xb   = (bf16*)alloc((size_t)MROWS * D_ * 2);
  bf16* wqb  = (bf16*)alloc((size_t)(H_ * HD_) * D_ * 2);
  bf16* wkb  = (bf16*)alloc((size_t)(KV_ * HD_) * D_ * 2);
  bf16* wvb  = (bf16*)alloc((size_t)(KV_ * HD_) * D_ * 2);
  bf16* wob  = (bf16*)alloc((size_t)D_ * (H_ * HD_) * 2);
  bf16* qraw = (bf16*)alloc((size_t)MROWS * H_ * HD_ * 2);
  bf16* kraw = (bf16*)alloc((size_t)MROWS * KV_ * HD_ * 2);
  bf16* vraw = (bf16*)alloc((size_t)MROWS * KV_ * HD_ * 2);
  bf16* qt   = (bf16*)alloc((size_t)MROWS * H_ * HD_ * 2);
  bf16* kt   = (bf16*)alloc((size_t)MROWS * KV_ * HD_ * 2);
  bf16* vt   = (bf16*)alloc((size_t)MROWS * KV_ * HD_ * 2);
  bf16* attn = (bf16*)alloc((size_t)MROWS * H_ * HD_ * 2);

  // 1) fp32 -> bf16
  auto cvt = [&](const float* in, bf16* out, size_t n) {
    int n4 = (int)(n / 4);
    cvt_f32_bf16<<<(n4 + 255) / 256, 256, 0, stream>>>(in, out, n4);
  };
  cvt(x,    xb,  (size_t)MROWS * D_);
  cvt(wq_w, wqb, (size_t)(H_ * HD_) * D_);
  cvt(wk_w, wkb, (size_t)(KV_ * HD_) * D_);
  cvt(wv_w, wvb, (size_t)(KV_ * HD_) * D_);
  cvt(wo_w, wob, (size_t)D_ * (H_ * HD_));

  // 2) QKV projections (bf16 WMMA GEMMs, +bias, bf16 out)
  gemm_bf16_wmma<true, true><<<dim3((H_ * HD_) / 64, MROWS / 128), 256, 0, stream>>>(
      xb, wqb, wq_b, qraw, MROWS, H_ * HD_, D_);
  gemm_bf16_wmma<true, true><<<dim3((KV_ * HD_) / 64, MROWS / 128), 256, 0, stream>>>(
      xb, wkb, wk_b, kraw, MROWS, KV_ * HD_, D_);
  gemm_bf16_wmma<true, true><<<dim3((KV_ * HD_) / 64, MROWS / 128), 256, 0, stream>>>(
      xb, wvb, wv_b, vraw, MROWS, KV_ * HD_, D_);

  // 3) RoPE (q, k) + V relayout to head-major
  rope_kernel<<<(B_ * S_ * H_ * 64) / 256, 256, 0, stream>>>(
      qraw, qt, fcos, fsin, spos, H_);
  rope_kernel<<<(B_ * S_ * KV_ * 64) / 256, 256, 0, stream>>>(
      kraw, kt, fcos, fsin, spos, KV_);
  vtrans_kernel<<<(B_ * S_ * KV_ * HD_) / 256, 256, 0, stream>>>(vraw, vt);

  // 4) causal flash attention
  flash_attn<<<dim3(S_ / 128, B_, H_), 256, 0, stream>>>(qt, kt, vt, attn);

  // 5) output projection -> fp32 d_out (no bias)
  gemm_bf16_wmma<false, false><<<dim3(D_ / 64, MROWS / 128), 256, 0, stream>>>(
      attn, wob, nullptr, (float*)d_out, MROWS, D_, H_ * HD_);
}
